// StackedLSTM_61795989454957
// MI455X (gfx1250) — compile-verified, matrix operand order checked by reading, not run on hardware
//
#include <hip/hip_runtime.h>
#include <hip/hip_bf16.h>

#define USE_ASYNC_LDS 1   // global_load_async_to_lds_b128 + s_wait_asynccnt
#define USE_TDM       1   // tensor_load_to_lds for the big W2 tile

// ---------- WMMA types (gfx1250 / CDNA5, wave32) ----------
typedef __attribute__((ext_vector_type(16))) __bf16 v16bf;
typedef __attribute__((ext_vector_type(8)))  float  v8f;

union FragU {
    uint4  q[2];   // 32 bytes
    v16bf  v;      // 16 x bf16
};

__device__ __forceinline__ v16bf load_frag16(const unsigned short* p) {
    FragU f;
    f.q[0] = *reinterpret_cast<const uint4*>(p);
    f.q[1] = *reinterpret_cast<const uint4*>(p + 8);
    return f.v;
}

__device__ __forceinline__ unsigned short f32_to_bf16_rne(float x) {
    unsigned int u = __float_as_uint(x);
    unsigned int r = u + 0x7FFFu + ((u >> 16) & 1u);
    return (unsigned short)(r >> 16);
}

// Cooperative async stage of `nElems` bf16 (multiple of 8) from contiguous
// global `src` into LDS at `dst`, 16B per lane per issue (ASYNCcnt-tracked).
__device__ __forceinline__ void stage_tile_async(const unsigned short* __restrict__ src,
                                                 unsigned short* dst, int nElems, int tid) {
    const int nChunks = nElems >> 3;   // 16B chunks
#if USE_ASYNC_LDS
    const unsigned long long gbase = (unsigned long long)src;
    for (int i = tid; i < nChunks; i += 256) {
        unsigned ldsOff = (unsigned)(uintptr_t)(dst + (size_t)i * 8);
        unsigned gOff   = (unsigned)(i * 16);
        asm volatile("global_load_async_to_lds_b128 %0, %1, %2"
                     :: "v"(ldsOff), "v"(gOff), "s"(gbase)
                     : "memory");
    }
#else
    for (int i = tid; i < nChunks; i += 256) {
        uint4 d = *reinterpret_cast<const uint4*>(src + (size_t)i * 8);
        *reinterpret_cast<uint4*>(dst + (size_t)i * 8) = d;
    }
#endif
}

#if USE_TDM
typedef __attribute__((ext_vector_type(4))) unsigned int u32x4;
typedef __attribute__((ext_vector_type(8))) int          i32x8;
typedef __attribute__((ext_vector_type(4))) int          i32x4;

// One 1D TDM DMA: contiguous copy of numElems8B 8-byte elements global -> LDS.
// D# built per CDNA5 ISA 8.3/8.4: count=1, type=2(image), data_size=3 (8B),
// tensor_dim0 = tile_dim0 = numElems8B, tensor_dim1 = tile_dim1 = 1.
__device__ __forceinline__ void tdm_load_1d(const void* gsrc, unsigned ldsAddr,
                                            unsigned numElems8B) {
    const unsigned long long ga = (unsigned long long)gsrc;
    u32x4 g0;
    g0[0] = 1u;                                               // count=1 (user)
    g0[1] = ldsAddr;                                          // lds_addr
    g0[2] = (unsigned)(ga & 0xFFFFFFFFu);                     // global_addr[31:0]
    g0[3] = (unsigned)((ga >> 32) & 0x01FFFFFFu) | (2u << 30);// [56:32] | type=2
    i32x8 g1;
    g1[0] = (int)(3u << 16);                                  // data_size=3 (8B)
    g1[1] = (int)((numElems8B & 0xFFFFu) << 16);              // tensor_dim0[15:0]
    g1[2] = (int)((numElems8B >> 16) | (1u << 16));           // td0[31:16] | td1.lo=1
    g1[3] = (int)((numElems8B & 0xFFFFu) << 16);              // td1.hi=0 | tile_dim0
    g1[4] = 1;                                                // tile_dim1=1, tile_dim2=0
    g1[5] = (int)numElems8B;                                  // tensor_dim0_stride lo
    g1[6] = 0;
    g1[7] = 0;
    i32x4 z4 = {0, 0, 0, 0};
#if __clang_major__ >= 23
    i32x8 z8 = {0, 0, 0, 0, 0, 0, 0, 0};
    __builtin_amdgcn_tensor_load_to_lds(g0, g1, z4, z4, z8, 0);
#else
    __builtin_amdgcn_tensor_load_to_lds(g0, g1, z4, z4, 0);
#endif
}
#endif

// ---------- fp32 -> bf16 conversion ----------
__global__ void __launch_bounds__(256) cvt_f32_bf16(const float* __restrict__ src,
                                                    unsigned short* __restrict__ dst, int n) {
    int i = blockIdx.x * blockDim.x + threadIdx.x;
    if (i < n) dst[i] = f32_to_bf16_rne(src[i]);
}

// ---------- Fused two-term gate GEMM, LDS-staged weights, 2 col-tiles/wave ----
// gates[B=128, G] = A1[128,K1] @ W1[G,K1]^T + A2[128,K2] @ W2[G,K2]^T + b_ih + b_hh
// Block: 32 gate columns x 128 batch rows. Wave w owns row-tile w; each wave
// computes two 16x16 tiles (cols n0 / n0+16), reusing one A fragment for two
// WMMAs (halves global A traffic, two independent WMMA chains).
// W1 tile staged via async global->LDS; W2 tile via one TDM descriptor.
// Dynamic LDS: 32*(K1+K2)*2 bytes.
__global__ void __launch_bounds__(256) lstm_gates_wmma(
    const unsigned short* __restrict__ A1, int lda1, int K1,
    const unsigned short* __restrict__ W1,
    const unsigned short* __restrict__ A2, int lda2, int K2,
    const unsigned short* __restrict__ W2,
    const float* __restrict__ b_ih, const float* __restrict__ b_hh,
    float* __restrict__ gates, int G)
{
    extern __shared__ unsigned short sW[];
    unsigned short* sW1 = sW;                      // 32 x K1
    unsigned short* sW2 = sW + (size_t)32 * K1;    // 32 x K2

    const int tid  = threadIdx.x;
    const int wave = tid >> 5;
    const int lane = tid & 31;
    const int n0   = blockIdx.x * 32;   // first gate column of block
    const int m0   = wave * 16;         // batch-row tile (B=128 -> 8 waves)
    const int half = lane >> 4;         // K half-selector (same perm for A and B)
    const int l    = lane & 15;         // row (A) / column (B) within tile

    // Stage weights: W1 rows via per-lane async copies, W2 rows via TDM DMA.
    stage_tile_async(W1 + (size_t)n0 * K1, sW1, 32 * K1, tid);
#if USE_TDM
    if (wave == 0) {
        tdm_load_1d(W2 + (size_t)n0 * K2, (unsigned)(uintptr_t)sW2,
                    (unsigned)(32 * K2 * 2 / 8));
        __builtin_amdgcn_s_wait_tensorcnt(0);
    }
#else
    stage_tile_async(W2 + (size_t)n0 * K2, sW2, 32 * K2, tid);
#endif
#if USE_ASYNC_LDS
    asm volatile("s_wait_asynccnt 0x0" ::: "memory");
#endif
    __syncthreads();

    v8f acc0 = {}, acc1 = {};

    // term 1: A1 @ W1^T   (layer1: x_t with K1=32 -> one WMMA per col-tile)
    {
        const unsigned short* ap  = A1  + (size_t)(m0 + l) * lda1 + half * 16;
        const unsigned short* w0p = sW1 + (size_t)l        * K1  + half * 16;
        const unsigned short* w1p = sW1 + (size_t)(l + 16) * K1  + half * 16;
        for (int k = 0; k < K1; k += 32) {
            v16bf a  = load_frag16(ap  + k);
            v16bf b0 = load_frag16(w0p + k);
            v16bf b1 = load_frag16(w1p + k);
            acc0 = __builtin_amdgcn_wmma_f32_16x16x32_bf16(
                false, a, false, b0, (short)0, acc0, false, false);
            acc1 = __builtin_amdgcn_wmma_f32_16x16x32_bf16(
                false, a, false, b1, (short)0, acc1, false, false);
        }
    }
    // term 2: A2 @ W2^T   (recurrent h term)
    {
        const unsigned short* ap  = A2  + (size_t)(m0 + l) * lda2 + half * 16;
        const unsigned short* w0p = sW2 + (size_t)l        * K2  + half * 16;
        const unsigned short* w1p = sW2 + (size_t)(l + 16) * K2  + half * 16;
        for (int k = 0; k < K2; k += 32) {
            v16bf a  = load_frag16(ap  + k);
            v16bf b0 = load_frag16(w0p + k);
            v16bf b1 = load_frag16(w1p + k);
            acc0 = __builtin_amdgcn_wmma_f32_16x16x32_bf16(
                false, a, false, b0, (short)0, acc0, false, false);
            acc1 = __builtin_amdgcn_wmma_f32_16x16x32_bf16(
                false, a, false, b1, (short)0, acc1, false, false);
        }
    }

    // Epilogue: C/D layout -> VGPR r, lanes 0-15: M=r, lanes 16-31: M=r+8; N=lane&15
    const float bias0 = b_ih[n0 + l]      + b_hh[n0 + l];
    const float bias1 = b_ih[n0 + 16 + l] + b_hh[n0 + 16 + l];
    float* gp0 = gates + (size_t)(m0 + half * 8) * G + n0 + l;
    float* gp1 = gp0 + 16;
#pragma unroll
    for (int r = 0; r < 8; ++r) {
        gp0[(size_t)r * G] = acc0[r] + bias0;
        gp1[(size_t)r * G] = acc1[r] + bias1;
    }
}

// ---------- LSTM pointwise: gate activations + state update ----------
__global__ void __launch_bounds__(256) lstm_pointwise(
    const float* __restrict__ gates, int H, int B,
    float* __restrict__ c,
    unsigned short* __restrict__ h_bf,   // bf16 h for next-step GEMMs
    float* __restrict__ h_f)             // optional fp32 h (for final head)
{
    int idx = blockIdx.x * blockDim.x + threadIdx.x;
    if (idx >= B * H) return;
    int b = idx / H;
    int j = idx - b * H;
    const float* g = gates + (size_t)b * 4 * H;
    float gi = g[j];
    float gf = g[H + j];
    float gg = g[2 * H + j];
    float go = g[3 * H + j];
    float si = 1.0f / (1.0f + __expf(-gi));
    float sf = 1.0f / (1.0f + __expf(-gf));
    float so = 1.0f / (1.0f + __expf(-go));
    float cn = sf * c[idx] + si * tanhf(gg);
    float hn = so * tanhf(cn);
    c[idx]    = cn;
    h_bf[idx] = f32_to_bf16_rne(hn);
    if (h_f) h_f[idx] = hn;
}

// ---------- Final linear head: out[b] = h2[b,:] . W_lin + b_lin ----------
__global__ void __launch_bounds__(256) final_linear(
    const float* __restrict__ h2, const float* __restrict__ W,
    const float* __restrict__ bias, float* __restrict__ out, int H)
{
    __shared__ float s[8];
    const int b = blockIdx.x;
    float sum = 0.f;
    for (int j = threadIdx.x; j < H; j += blockDim.x)
        sum += h2[(size_t)b * H + j] * W[j];
    for (int off = 16; off > 0; off >>= 1)
        sum += __shfl_down(sum, off, 32);
    if ((threadIdx.x & 31) == 0) s[threadIdx.x >> 5] = sum;
    __syncthreads();
    if (threadIdx.x == 0) {
        float t = 0.f;
#pragma unroll
        for (int w = 0; w < 8; ++w) t += s[w];
        out[b] = t + bias[0];
    }
}

extern "C" void kernel_launch(void* const* d_in, const int* in_sizes, int n_in,
                              void* d_out, int out_size, void* d_ws, size_t ws_size,
                              hipStream_t stream)
{
    const float* x     = (const float*)d_in[0];
    const float* W_ih1 = (const float*)d_in[1];
    const float* W_hh1 = (const float*)d_in[2];
    const float* b_ih1 = (const float*)d_in[3];
    const float* b_hh1 = (const float*)d_in[4];
    const float* W_ih2 = (const float*)d_in[5];
    const float* W_hh2 = (const float*)d_in[6];
    const float* b_ih2 = (const float*)d_in[7];
    const float* b_hh2 = (const float*)d_in[8];
    const float* W_lin = (const float*)d_in[9];
    const float* b_lin = (const float*)d_in[10];

    constexpr int B = 128, T = 256, I = 32, H1 = 512, H2 = 1024;
    constexpr int G1 = 4 * H1, G2 = 4 * H2;

    // Workspace carve-out (~21.7 MB total), 256B-aligned chunks.
    char*  ws  = (char*)d_ws;
    size_t off = 0;
    auto alloc = [&](size_t bytes) -> void* {
        void* p = ws + off;
        off = (off + bytes + 255) & ~size_t(255);
        return p;
    };
    unsigned short* xbf  = (unsigned short*)alloc((size_t)B * T * I * 2);
    unsigned short* wih1 = (unsigned short*)alloc((size_t)G1 * I  * 2);
    unsigned short* whh1 = (unsigned short*)alloc((size_t)G1 * H1 * 2);
    unsigned short* wih2 = (unsigned short*)alloc((size_t)G2 * H1 * 2);
    unsigned short* whh2 = (unsigned short*)alloc((size_t)G2 * H2 * 2);
    unsigned short* h1bf = (unsigned short*)alloc((size_t)B * H1 * 2);
    unsigned short* h2bf = (unsigned short*)alloc((size_t)B * H2 * 2);
    float* c1     = (float*)alloc((size_t)B * H1 * 4);
    float* c2     = (float*)alloc((size_t)B * H2 * 4);
    float* h2f    = (float*)alloc((size_t)B * H2 * 4);
    float* gates1 = (float*)alloc((size_t)B * G1 * 4);
    float* gates2 = (float*)alloc((size_t)B * G2 * 4);

    // Zero initial state (capture-legal async memsets).
    hipMemsetAsync(h1bf, 0, (size_t)B * H1 * 2, stream);
    hipMemsetAsync(h2bf, 0, (size_t)B * H2 * 2, stream);
    hipMemsetAsync(c1,   0, (size_t)B * H1 * 4, stream);
    hipMemsetAsync(c2,   0, (size_t)B * H2 * 4, stream);

    auto cvt = [&](const float* s, unsigned short* d, int n) {
        cvt_f32_bf16<<<(n + 255) / 256, 256, 0, stream>>>(s, d, n);
    };
    cvt(x,     xbf,  B * T * I);
    cvt(W_ih1, wih1, G1 * I);
    cvt(W_hh1, whh1, G1 * H1);
    cvt(W_ih2, wih2, G2 * H1);
    cvt(W_hh2, whh2, G2 * H2);

    const size_t lds1 = (size_t)32 * (I  + H1) * sizeof(unsigned short); // 34.8 KB
    const size_t lds2 = (size_t)32 * (H1 + H2) * sizeof(unsigned short); // 96 KB

    for (int t = 0; t < T; ++t) {
        // Layer 1: gates1 = x_t @ W_ih1^T + h1 @ W_hh1^T + b
        lstm_gates_wmma<<<G1 / 32, 256, lds1, stream>>>(
            xbf + (size_t)t * I, T * I, I, wih1,
            h1bf, H1, H1, whh1,
            b_ih1, b_hh1, gates1, G1);
        lstm_pointwise<<<(B * H1 + 255) / 256, 256, 0, stream>>>(
            gates1, H1, B, c1, h1bf, nullptr);

        // Layer 2: gates2 = h1 @ W_ih2^T + h2 @ W_hh2^T + b
        lstm_gates_wmma<<<G2 / 32, 256, lds2, stream>>>(
            h1bf, H1, H1, wih2,
            h2bf, H2, H2, whh2,
            b_ih2, b_hh2, gates2, G2);
        lstm_pointwise<<<(B * H2 + 255) / 256, 256, 0, stream>>>(
            gates2, H2, B, c2, h2bf, h2f);
    }

    final_linear<<<B, 256, 0, stream>>>(h2f, W_lin, b_lin, (float*)d_out, H2);
}
